// AttentionLSTMDecoder_68788196213302
// MI455X (gfx1250) — matrix operand order repaired and checked by reading, not compiled, not run on hardware
//
#include <hip/hip_runtime.h>
#include <hip/hip_bf16.h>
#include <stdint.h>

// ---------------- problem constants ----------------
#define BB   16
#define TT   64
#define SS   256
#define HH   512
#define EE   512
#define ENCC 512
#define VV   32000
#define G4H  2048   // 4*H

#define LSTR 40     // padded LDS row stride (ushorts) for the staged fallback

typedef __attribute__((ext_vector_type(16))) __bf16 v16bf;
typedef __attribute__((ext_vector_type(8)))  float  v8f;

union Frag {
    uint4  q[2];
    v16bf  v;
};

__device__ __forceinline__ unsigned short f2bf(float f) {
    unsigned int u = __float_as_uint(f);
    u = u + 0x7FFFu + ((u >> 16) & 1u);   // round-to-nearest-even
    return (unsigned short)(u >> 16);
}

__device__ __forceinline__ float sigmoidf(float x) {
    return 1.0f / (1.0f + __expf(-x));
}

// ---------------- converts ----------------
__global__ void f32_to_bf16_kernel(const float* __restrict__ src,
                                   unsigned short* __restrict__ dst, int n) {
    int i = blockIdx.x * blockDim.x + threadIdx.x;
    if (i < n) dst[i] = f2bf(src[i]);
}

// W [K, N] f32 row-major -> Wt [N, K] bf16 (2D grid: x over N, y over K)
__global__ void transpose_bf16_kernel(const float* __restrict__ W,
                                      unsigned short* __restrict__ Wt,
                                      int K, int N) {
    int n = blockIdx.x * blockDim.x + threadIdx.x;
    int k = blockIdx.y;
    if (n < N) Wt[(size_t)n * K + k] = f2bf(W[(size_t)k * N + n]);
}

// ---------------- direct bf16 WMMA GEMM (no LDS) ----------------
// C[M,N] = A[M,K](bf16 row-major) @ Bt[N,K](bf16, B transposed) + bias
// block = 256 thr (8 waves), tile 32(M) x 64(N); every fragment is
// 2x global_load_b128 of contiguous k (k = kc + 16*hi + s, same order for A and B).
__global__ void gemm_bf16_direct(const unsigned short* __restrict__ A,
                                 const unsigned short* __restrict__ Bt,
                                 const float* __restrict__ bias,
                                 float* __restrict__ C,
                                 int M, int N, int K) {
    const int tid  = threadIdx.x;
    const int lane = tid & 31;
    const int hi   = lane >> 4;
    const int ln   = lane & 15;
    const int w    = tid >> 5;
    const int wm   = w >> 2;           // 0..1
    const int wn   = w & 3;            // 0..3
    const int n0   = blockIdx.x * 64;
    const int m0   = blockIdx.y * 32;

    const int row = m0 + wm * 16 + ln;
    const int col = n0 + wn * 16 + ln;

    const unsigned short* ap = A  + (size_t)row * K;
    const unsigned short* bp = Bt + (size_t)col * K;

    v8f acc = {0.f,0.f,0.f,0.f,0.f,0.f,0.f,0.f};

    #pragma unroll 2
    for (int kc = 0; kc < K; kc += 32) {
        Frag fa, fb;
        fa.q[0] = *(const uint4*)(ap + kc + 16 * hi);         // k = kc+16hi .. +7
        fa.q[1] = *(const uint4*)(ap + kc + 16 * hi + 8);     // k = kc+16hi+8 .. +15
        const uint4* b4 = (const uint4*)(bp + kc + 16 * hi);  // k = kc+16hi .. +15
        fb.q[0] = b4[0];
        fb.q[1] = b4[1];
        acc = __builtin_amdgcn_wmma_f32_16x16x32_bf16(false, fa.v, false, fb.v,
                                                      (short)0, acc, false, false);
    }

    const float bb = bias ? bias[col] : 0.0f;
    #pragma unroll
    for (int r = 0; r < 8; ++r) {
        int m = m0 + wm * 16 + r + hi * 8;   // C/D layout (ISA 7.12.2)
        C[(size_t)m * N + col] = acc[r] + bb;
    }
}

// ---------------- staged fallback GEMM (A bf16, B f32 converted in-kernel) ----------------
__global__ void gemm_wmma_staged(const unsigned short* __restrict__ A,
                                 const float* __restrict__ Bm,
                                 const float* __restrict__ bias,
                                 float* __restrict__ C,
                                 int M, int N, int K) {
    const int tid  = threadIdx.x;
    const int lane = tid & 31;
    const int hi   = lane >> 4;
    const int ln   = lane & 15;
    const int w    = tid >> 5;
    const int wm   = w >> 2;
    const int wn   = w & 3;
    const int n0   = blockIdx.x * 64;
    const int m0   = blockIdx.y * 32;

    __shared__ unsigned short sBt[64 * LSTR];  // [col][k] transposed tile

    const int row = m0 + wm * 16 + ln;
    const int bcol = wn * 16 + ln;
    const unsigned short* ap = A + (size_t)row * K;

    v8f acc = {0.f,0.f,0.f,0.f,0.f,0.f,0.f,0.f};

    for (int kc = 0; kc < K; kc += 32) {
        __syncthreads();
        for (int i = tid; i < 2048; i += 256) {          // B tile 32x64 -> transposed
            int r = i >> 6, c = i & 63;
            sBt[c * LSTR + r] = f2bf(Bm[(size_t)(kc + r) * N + n0 + c]);
        }
        if (kc + 32 < K && tid < 32)
            __builtin_prefetch(Bm + (size_t)(kc + 32 + tid) * N + n0, 0, 1);
        __syncthreads();

        Frag fa, fb;
        fa.q[0] = *(const uint4*)(ap + kc + 16 * hi);
        fa.q[1] = *(const uint4*)(ap + kc + 16 * hi + 8);
        fb.q[0] = *(const uint4*)&sBt[bcol * LSTR + 16 * hi];
        fb.q[1] = *(const uint4*)&sBt[bcol * LSTR + 16 * hi + 8];

        acc = __builtin_amdgcn_wmma_f32_16x16x32_bf16(false, fa.v, false, fb.v,
                                                      (short)0, acc, false, false);
    }

    const int n = n0 + bcol;
    const float bb = bias ? bias[n] : 0.0f;
    #pragma unroll
    for (int r = 0; r < 8; ++r) {
        int m = m0 + wm * 16 + r + hi * 8;
        C[(size_t)m * N + n] = acc[r] + bb;
    }
}

// ---------------- per-step attention (B blocks of 256 threads) ----------------
__global__ void attention_step_kernel(const int* __restrict__ input_ids,      // [B,T]
                                      const float* __restrict__ embedding,    // [V,E]
                                      const float* __restrict__ keys,         // [B,S,H]
                                      const float* __restrict__ enc,          // [B,S,ENC]
                                      const int* __restrict__ enc_mask,       // [B,S]
                                      const float* __restrict__ Wq,           // [H,H]
                                      const float* __restrict__ We,           // [H]
                                      const float* __restrict__ h1,           // [B,H] f32
                                      unsigned short* __restrict__ xcat,      // [B,1024] bf16
                                      unsigned short* __restrict__ outcat,    // [B*T,1024] bf16
                                      float* __restrict__ attn_out,           // [B,T,S]
                                      int t) {
    const int b = blockIdx.x;
    const int tid = threadIdx.x;
    __shared__ float h1s[HH];
    __shared__ float qs[HH];
    __shared__ float ss[SS];
    __shared__ float red[SS];

    h1s[tid]       = h1[b * HH + tid];
    h1s[tid + 256] = h1[b * HH + 256 + tid];
    __syncthreads();

    // q = h1 @ Wq
    for (int j = tid; j < HH; j += 256) {
        float a = 0.f;
        for (int k = 0; k < HH; ++k) a += h1s[k] * Wq[k * HH + j];
        qs[j] = a;
    }
    __syncthreads();

    // scores[s] = (We . tanh(q + keys[b,s,:])) * scale, masked
    {
        const int s = tid;
        float sc;
        if (enc_mask[b * SS + s] == 0) sc = -1e9f;
        else {
            const float* kp = keys + ((size_t)b * SS + s) * HH;
            float a = 0.f;
            for (int h = 0; h < HH; ++h) a += We[h] * tanhf(qs[h] + kp[h]);
            sc = a * 0.04419417382415922f;               // 1/sqrt(512)
        }
        ss[tid] = sc;
        red[tid] = sc;
    }
    __syncthreads();
    for (int off = 128; off > 0; off >>= 1) {
        if (tid < off) red[tid] = fmaxf(red[tid], red[tid + off]);
        __syncthreads();
    }
    const float mx = red[0];
    __syncthreads();
    const float ex = __expf(ss[tid] - mx);
    red[tid] = ex;
    __syncthreads();
    for (int off = 128; off > 0; off >>= 1) {
        if (tid < off) red[tid] += red[tid + off];
        __syncthreads();
    }
    const float a = ex / red[0];
    ss[tid] = a;
    attn_out[((size_t)b * TT + t) * SS + tid] = a;
    __syncthreads();

    // context[j] = sum_s attn[s] * enc[b,s,j]  -> bf16 into xcat & outcat
    for (int j = tid; j < ENCC; j += 256) {
        float c = 0.f;
        const float* ep = enc + (size_t)b * SS * ENCC + j;
        for (int s2 = 0; s2 < SS; ++s2) c += ss[s2] * ep[(size_t)s2 * ENCC];
        unsigned short cb = f2bf(c);
        xcat[b * 1024 + 512 + j] = cb;
        outcat[((size_t)b * TT + t) * 1024 + 512 + j] = cb;
    }
    // embedding gather -> xcat[:,0:512]
    const int tok = input_ids[b * TT + t];
    for (int j = tid; j < EE; j += 256)
        xcat[b * 1024 + j] = f2bf(embedding[(size_t)tok * EE + j]);
}

// ---------------- fused LSTM cell step (no LDS, no barriers) ----------------
// gates = A[16,K1]@W1 + Hp[16,512]@W2 + bih + bhh; weights pre-transposed bf16
// (Wt[col][k]); A rows are 16B-aligned bf16 so both fragments come straight
// from global as b128 clauses. Each wave owns 16 hidden units, 4 accumulators
// = i/f/g/o columns, nonlinearity closes in-register. 8 blocks x 128 thr.
__global__ void lstm_cell_kernel(const unsigned short* __restrict__ A, int K1,
                                 const unsigned short* __restrict__ W1t,  // [2048, K1]
                                 const unsigned short* __restrict__ Hp,
                                 const unsigned short* __restrict__ W2t,  // [2048, 512]
                                 const float* __restrict__ bih,
                                 const float* __restrict__ bhh,
                                 float* __restrict__ c_io,
                                 float* __restrict__ h_out,
                                 unsigned short* __restrict__ hb_out,
                                 unsigned short* __restrict__ outcat,
                                 int t) {
    const int tid  = threadIdx.x;
    const int lane = tid & 31;
    const int hi   = lane >> 4;
    const int ln   = lane & 15;
    const int wid  = blockIdx.x * 4 + (tid >> 5);
    const int n0   = wid * 16;

    v8f acc[4];
    #pragma unroll
    for (int g = 0; g < 4; ++g)
        #pragma unroll
        for (int r = 0; r < 8; ++r) acc[g][r] = 0.0f;

    for (int phase = 0; phase < 2; ++phase) {
        const unsigned short* Ap = phase ? Hp : A;
        const unsigned short* Wt = phase ? W2t : W1t;
        const int KK = phase ? HH : K1;
        const unsigned short* arow = Ap + (size_t)ln * KK;   // A row m = ln
        for (int kc = 0; kc < KK; kc += 32) {
            Frag fa;
            fa.q[0] = *(const uint4*)(arow + kc + 16 * hi);
            fa.q[1] = *(const uint4*)(arow + kc + 16 * hi + 8);
            #pragma unroll
            for (int g = 0; g < 4; ++g) {
                const int col = g * HH + n0 + ln;
                const uint4* wp = (const uint4*)(Wt + (size_t)col * KK + kc + 16 * hi);
                Frag fb;
                fb.q[0] = wp[0];                          // 32 contiguous bytes:
                fb.q[1] = wp[1];                          // k = kc+16hi .. +15
                acc[g] = __builtin_amdgcn_wmma_f32_16x16x32_bf16(
                    false, fa.v, false, fb.v, (short)0, acc[g], false, false);
            }
        }
    }

    // epilogue: torch gate order i,f,g,o along 4H
    const int n = n0 + ln;
    const float bi = bih[n]        + bhh[n];
    const float bf = bih[512 + n]  + bhh[512 + n];
    const float bg = bih[1024 + n] + bhh[1024 + n];
    const float bo = bih[1536 + n] + bhh[1536 + n];
    #pragma unroll
    for (int r = 0; r < 8; ++r) {
        const int m = r + hi * 8;                        // batch row
        float iv = sigmoidf(acc[0][r] + bi);
        float fv = sigmoidf(acc[1][r] + bf);
        float gv = tanhf(acc[2][r] + bg);
        float ov = sigmoidf(acc[3][r] + bo);
        float cn = fv * c_io[m * HH + n] + iv * gv;
        float hn = ov * tanhf(cn);
        c_io[m * HH + n]  = cn;
        h_out[m * HH + n] = hn;
        unsigned short hb = f2bf(hn);
        hb_out[m * HH + n] = hb;
        if (outcat) outcat[((size_t)m * TT + t) * 1024 + n] = hb;
    }
}

// ---------------- launch ----------------
extern "C" void kernel_launch(void* const* d_in, const int* in_sizes, int n_in,
                              void* d_out, int out_size, void* d_ws, size_t ws_size,
                              hipStream_t stream) {
    const int*   input_ids = (const int*)  d_in[0];
    const float* hidden    = (const float*)d_in[1];   // [2,16,512]
    const float* cellin    = (const float*)d_in[2];   // [2,16,512]
    const float* enc       = (const float*)d_in[3];   // [16,256,512]
    const int*   enc_mask  = (const int*)  d_in[4];
    const float* embedding = (const float*)d_in[5];
    const float* Wq        = (const float*)d_in[6];
    const float* Wk        = (const float*)d_in[7];
    const float* We        = (const float*)d_in[8];
    const float* W_ih0     = (const float*)d_in[9];   // [1024,2048]
    const float* W_hh0     = (const float*)d_in[10];  // [512,2048]
    const float* b_ih0     = (const float*)d_in[11];
    const float* b_hh0     = (const float*)d_in[12];
    const float* W_ih1     = (const float*)d_in[13];  // [512,2048]
    const float* W_hh1     = (const float*)d_in[14];
    const float* b_ih1     = (const float*)d_in[15];
    const float* b_hh1     = (const float*)d_in[16];
    const float* W_out     = (const float*)d_in[17];  // [1024,32000]
    const float* b_out     = (const float*)d_in[18];

    // output layout: predictions [B,T,V], h_f [2,B,H], c_f [2,B,H], attn [B,T,S]
    float* pred = (float*)d_out;
    float* hf   = pred + (size_t)BB * TT * VV;
    float* cf   = hf + 2 * BB * HH;
    float* attn = cf + 2 * BB * HH;

    // workspace carve (256B aligned)
    uintptr_t p = (uintptr_t)d_ws;
    auto carve = [&](size_t bytes) -> void* {
        void* r = (void*)p;
        p += (bytes + 255) & ~(size_t)255;
        return r;
    };
    float*          keys   = (float*)         carve((size_t)BB * SS * HH * 4);
    unsigned short* wih0t  = (unsigned short*)carve((size_t)G4H * 1024 * 2);  // [2048,1024]
    unsigned short* whh0t  = (unsigned short*)carve((size_t)G4H * HH * 2);    // [2048,512]
    unsigned short* wih1t  = (unsigned short*)carve((size_t)G4H * HH * 2);
    unsigned short* whh1t  = (unsigned short*)carve((size_t)G4H * HH * 2);
    unsigned short* encb   = (unsigned short*)carve((size_t)BB * SS * ENCC * 2); // bf16 enc
    unsigned short* wkt    = (unsigned short*)carve((size_t)HH * ENCC * 2);      // [H,ENC] = Wk^T
    float*          h_ws   = (float*)         carve((size_t)2 * BB * HH * 4);
    float*          c_ws   = (float*)         carve((size_t)2 * BB * HH * 4);
    unsigned short* hb0[2], *hb1[2];
    hb0[0] = (unsigned short*)carve((size_t)BB * HH * 2);
    hb0[1] = (unsigned short*)carve((size_t)BB * HH * 2);
    hb1[0] = (unsigned short*)carve((size_t)BB * HH * 2);
    hb1[1] = (unsigned short*)carve((size_t)BB * HH * 2);
    unsigned short* xcat   = (unsigned short*)carve((size_t)BB * 1024 * 2);
    unsigned short* outcat = (unsigned short*)carve((size_t)BB * TT * 1024 * 2);

    // optional: pre-transposed bf16 W_out [V,1024] (dominant GEMM, L2-resident)
    size_t used = p - (uintptr_t)d_ws;
    const size_t woutt_bytes = (size_t)VV * 1024 * 2;
    unsigned short* woutt = nullptr;
    if (ws_size >= used + woutt_bytes + 256)
        woutt = (unsigned short*)carve(woutt_bytes);

    // ---- init: weights -> transposed bf16, activations -> bf16, state copies ----
    transpose_bf16_kernel<<<dim3(G4H / 256, 1024), 256, 0, stream>>>(W_ih0, wih0t, 1024, G4H);
    transpose_bf16_kernel<<<dim3(G4H / 256, HH), 256, 0, stream>>>(W_hh0, whh0t, HH, G4H);
    transpose_bf16_kernel<<<dim3(G4H / 256, HH), 256, 0, stream>>>(W_ih1, wih1t, HH, G4H);
    transpose_bf16_kernel<<<dim3(G4H / 256, HH), 256, 0, stream>>>(W_hh1, whh1t, HH, G4H);
    transpose_bf16_kernel<<<dim3(HH / 256, ENCC), 256, 0, stream>>>(Wk, wkt, ENCC, HH);
    f32_to_bf16_kernel<<<(BB * SS * ENCC) / 256, 256, 0, stream>>>(enc, encb, BB * SS * ENCC);
    f32_to_bf16_kernel<<<32, 256, 0, stream>>>(hidden, hb0[0], BB * HH);
    f32_to_bf16_kernel<<<32, 256, 0, stream>>>(hidden + BB * HH, hb1[0], BB * HH);
    hipMemcpyAsync(h_ws, hidden, (size_t)2 * BB * HH * 4, hipMemcpyDeviceToDevice, stream);
    hipMemcpyAsync(c_ws, cellin, (size_t)2 * BB * HH * 4, hipMemcpyDeviceToDevice, stream);
    if (woutt)
        transpose_bf16_kernel<<<dim3(VV / 256 + 1, 1024), 256, 0, stream>>>(W_out, woutt, 1024, VV);

    // ---- keys = enc @ Wk : [4096,512] x [512,512] (direct bf16 WMMA) ----
    gemm_bf16_direct<<<dim3(HH / 64, (BB * SS) / 32), 256, 0, stream>>>(
        encb, wkt, nullptr, keys, BB * SS, HH, ENCC);

    // ---- recurrent loop ----
    for (int t = 0; t < TT; ++t) {
        attention_step_kernel<<<BB, 256, 0, stream>>>(
            input_ids, embedding, keys, enc, enc_mask, Wq, We,
            h_ws + BB * HH /* layer-1 h */, xcat, outcat, attn, t);
        lstm_cell_kernel<<<8, 128, 0, stream>>>(
            xcat, 1024, wih0t, hb0[t & 1], whh0t, b_ih0, b_hh0,
            c_ws, h_ws, hb0[(t + 1) & 1], nullptr, t);
        lstm_cell_kernel<<<8, 128, 0, stream>>>(
            hb0[(t + 1) & 1], HH, wih1t, hb1[t & 1], whh1t, b_ih1, b_hh1,
            c_ws + BB * HH, h_ws + BB * HH, hb1[(t + 1) & 1], outcat, t);
    }

    // ---- predictions = outcat @ W_out + b_out : [1024,1024] x [1024,32000] ----
    if (woutt)
        gemm_bf16_direct<<<dim3(VV / 64, (BB * TT) / 32), 256, 0, stream>>>(
            outcat, woutt, b_out, pred, BB * TT, VV, 1024);
    else
        gemm_wmma_staged<<<dim3(VV / 64, (BB * TT) / 32), 256, 0, stream>>>(
            outcat, W_out, b_out, pred, BB * TT, VV, 1024);

    // ---- final state outputs ----
    hipMemcpyAsync(hf, h_ws, (size_t)2 * BB * HH * 4, hipMemcpyDeviceToDevice, stream);
    hipMemcpyAsync(cf, c_ws, (size_t)2 * BB * HH * 4, hipMemcpyDeviceToDevice, stream);
}
